// MultiHeadAttention_17188459118962
// MI455X (gfx1250) — matrix operand order repaired and checked by reading, not compiled, not run on hardware
//
#include <hip/hip_runtime.h>
#include <hip/hip_bf16.h>

// MHA forward for gfx1250 (MI455X). B=2,T=2048,D=1024,H=16,DK=64.
// bf16 WMMA for all 4 matmul stages; TDM (tensor_load_to_lds) double-buffered
// staging for the projection GEMMs; fused attention keeps the 16x2048 f32
// score strip in CDNA5's 320KB LDS so the 537MB attn matrix hits HBM once.

#define B_ 2
#define T_ 2048
#define D_ 1024
#define H_ 16
#define DK_ 64

typedef __attribute__((ext_vector_type(16))) __bf16 v16bf;
typedef __attribute__((ext_vector_type(8)))  float  v8f;
typedef __attribute__((ext_vector_type(4)))  unsigned int   u32x4;
typedef __attribute__((ext_vector_type(4)))  int            i32x4;
typedef __attribute__((ext_vector_type(8)))  int            i32x8;
typedef __attribute__((ext_vector_type(4)))  unsigned short us4;

static __device__ __forceinline__ unsigned short f2bf(float f) {
  union { float f; unsigned u; } v; v.f = f;
  unsigned r = v.u + 0x7FFFu + ((v.u >> 16) & 1u);   // round-to-nearest-even
  return (unsigned short)(r >> 16);
}
static __device__ __forceinline__ __bf16 us2bf(unsigned short u) {
  return __builtin_bit_cast(__bf16, u);
}
// A-fragment (16x32 bf16): lane l (M=l&15), element e -> K index:
static __device__ __forceinline__ int ka_idx(int e, int half) {
  return (e & 7) + ((e >> 3) << 4) + (half << 3);
}
// B-fragment (32x16 bf16): lane l (N=l&15), element e -> K index:
static __device__ __forceinline__ int kb_idx(int e, int half) {
  return (e & 7) + ((e >> 3) << 4) + (half << 3);
}
// Low 32 bits of a generic LDS pointer = byte offset within LDS allocation.
static __device__ __forceinline__ unsigned lds_off(const void* p) {
  return (unsigned)(unsigned long long)(size_t)p;
}

// Issue one TDM copy of a 64-row x 32-col bf16 tile (4KB), row stride 1024
// elements, from global tile start `gptr` into LDS at byte offset `loff`.
// D# group0: count=1, type=2(image); group1: data_size=1(2B), tile 32x64.
static __device__ __forceinline__ void tdm_load_tile(unsigned loff,
                                                     const unsigned short* gptr,
                                                     unsigned rows_avail) {
  unsigned long long ga = (unsigned long long)(size_t)gptr;
  u32x4 g0 = { 1u, loff, (unsigned)ga,
               (unsigned)((ga >> 32) & 0x1FFFFFFull) | 0x80000000u };
  i32x8 g1 = { 0x00010000,                       // data_size=1 (2-byte elems)
               0x04000000,                       // tensor_dim0=1024 (bits79:48)
               (int)((rows_avail & 0xFFFFu) << 16),   // tensor_dim1 lo
               (int)((rows_avail >> 16) | (32u << 16)), // dim1 hi | tile_dim0=32
               64,                               // tile_dim1 = 64 rows
               1024,                             // tensor_dim0_stride = 1024
               0, 0 };
  i32x4 z4 = { 0, 0, 0, 0 };
  i32x8 z8 = { 0, 0, 0, 0, 0, 0, 0, 0 };
  __builtin_amdgcn_tensor_load_to_lds(g0, g1, z4, z4, z8, 0);
}

// ---------------------------------------------------------------- convert
__global__ void cvt_f32_bf16(const float4* __restrict__ src,
                             us4* __restrict__ dst, int n4) {
  for (int i = blockIdx.x * blockDim.x + threadIdx.x; i < n4;
       i += gridDim.x * blockDim.x) {
    float4 v = src[i];
    us4 o = { f2bf(v.x), f2bf(v.y), f2bf(v.z), f2bf(v.w) };
    dst[i] = o;
  }
}

// ------------------------------------------------- GEMM: out = A @ W^T + b
// A bf16 [M][1024] row-major, W bf16 [N=1024][1024] row-major.
// mode 0: bf16 [B,H,T,DK] head layout (Q,K). mode 2: bf16 [B,H,DK,T]
// (V transposed for the PV WMMA). mode 1: f32 [M][N] (final projection).
__global__ __launch_bounds__(128)
void gemm_nt(const unsigned short* __restrict__ A,
             const unsigned short* __restrict__ W,
             const float* __restrict__ bias,
             unsigned short* __restrict__ out_head,
             float* __restrict__ out_f32, int mode, int m_rows) {
  __shared__ unsigned short As[2][64 * 32];   // double-buffered [row][k]
  __shared__ unsigned short Bs[2][64 * 32];   // double-buffered [n][k]
  const int row0 = blockIdx.y * 64, col0 = blockIdx.x * 64;
  const int tid = threadIdx.x, lane = tid & 31, w = tid >> 5;
  const int half = lane >> 4, l16 = lane & 15;

  v8f acc[4];
#pragma unroll
  for (int nt = 0; nt < 4; ++nt)
#pragma unroll
    for (int r = 0; r < 8; ++r) acc[nt][r] = 0.0f;

  const int ITER = D_ / 32;
  if (w == 0) {  // prologue: TDM fills buffer 0 (wave-uniform branch)
    tdm_load_tile(lds_off(&As[0][0]), A + (size_t)row0 * D_, (unsigned)m_rows);
    tdm_load_tile(lds_off(&Bs[0][0]), W + (size_t)col0 * D_, 1024u);
  }

  for (int i = 0; i < ITER; ++i) {
    const int buf = i & 1;
    if (w == 0) {
      if (i + 1 < ITER) {  // issue next tile, wait for current (2 newest pend)
        int k0 = (i + 1) * 32;
        tdm_load_tile(lds_off(&As[buf ^ 1][0]),
                      A + (size_t)row0 * D_ + k0, (unsigned)m_rows);
        tdm_load_tile(lds_off(&Bs[buf ^ 1][0]),
                      W + (size_t)col0 * D_ + k0, 1024u);
        __builtin_amdgcn_s_wait_tensorcnt(2);
      } else {
        __builtin_amdgcn_s_wait_tensorcnt(0);
      }
    }
    __syncthreads();   // TDM data visible to all waves

    v16bf a;
#pragma unroll
    for (int e = 0; e < 16; ++e)
      a[e] = us2bf(As[buf][(w * 16 + l16) * 32 + ka_idx(e, half)]);
#pragma unroll
    for (int nt = 0; nt < 4; ++nt) {
      v16bf b;
#pragma unroll
      for (int e = 0; e < 16; ++e)
        b[e] = us2bf(Bs[buf][(nt * 16 + l16) * 32 + kb_idx(e, half)]);
      acc[nt] = __builtin_amdgcn_wmma_f32_16x16x32_bf16(
          false, a, false, b, (short)0, acc[nt], false, false);
    }
    __syncthreads();   // all waves done with buf before TDM overwrites it
  }

#pragma unroll
  for (int nt = 0; nt < 4; ++nt)
#pragma unroll
    for (int r = 0; r < 8; ++r) {
      int m = row0 + w * 16 + r + 8 * half;
      int n = col0 + nt * 16 + l16;
      float v = acc[nt][r] + bias[n];
      if (mode == 0) {          // [B,H,T,DK] bf16
        int bb = m >> 11, t = m & (T_ - 1), h = n >> 6, dk = n & 63;
        out_head[(((size_t)bb * H_ + h) * T_ + t) * DK_ + dk] = f2bf(v);
      } else if (mode == 2) {   // [B,H,DK,T] bf16 (V transposed)
        int bb = m >> 11, t = m & (T_ - 1), h = n >> 6, dk = n & 63;
        out_head[(((size_t)bb * H_ + h) * DK_ + dk) * T_ + t] = f2bf(v);
      } else {                  // f32 [M][N]
        out_f32[(size_t)m * D_ + n] = v;
      }
    }
}

// ------------------------------------------------------- fused attention
// grid = (T/16, B*H). One workgroup: 16 query rows of one (b,h).
// LDS: 16x2048 f32 scores (128KB) + Q tile + split-K scratch (~146KB).
__global__ __launch_bounds__(256)
void attn_fused(const unsigned short* __restrict__ Qh,
                const unsigned short* __restrict__ Kh,
                const unsigned short* __restrict__ Vt,
                unsigned short* __restrict__ hbuf,
                float* __restrict__ attn_out) {
  __shared__ float sS[16 * T_];
  __shared__ unsigned short sQ[16 * DK_];
  __shared__ float sRed[4 * 16 * 64];

  const int qt = blockIdx.x, bh = blockIdx.y, q0 = qt * 16;
  const int tid = threadIdx.x, lane = tid & 31, w = tid >> 5;
  const int half = lane >> 4, l16 = lane & 15;
  const size_t base = (size_t)bh * T_ * DK_;   // same size for Kh and Vt

  {  // cooperative Q tile load (16x64 bf16 = 512 dwords)
    const unsigned int* Qw = (const unsigned int*)(Qh + base + (size_t)q0 * DK_);
    unsigned int* sQw = (unsigned int*)sQ;
    sQw[tid] = Qw[tid];
    sQw[tid + 256] = Qw[tid + 256];
  }
  __syncthreads();

  // ---- phase 1: S = (Q K^T) * 1/sqrt(DK), causal mask, into LDS ----
  const float scale = 0.125f;
  for (int kt = w; kt <= qt; kt += 8) {
    if (kt + 8 <= qt)   // warm next K tile for this wave
      __builtin_prefetch(Kh + base + (size_t)((kt + 8) * 16 + l16) * DK_, 0, 1);
    v8f acc;
#pragma unroll
    for (int r = 0; r < 8; ++r) acc[r] = 0.0f;
#pragma unroll
    for (int kh = 0; kh < 2; ++kh) {        // DK=64 -> two K=32 steps
      v16bf a, b;
#pragma unroll
      for (int e = 0; e < 16; ++e)
        a[e] = us2bf(sQ[l16 * DK_ + kh * 32 + ka_idx(e, half)]);
      const unsigned short* Kp =
          Kh + base + (size_t)(kt * 16 + l16) * DK_ + kh * 32;
#pragma unroll
      for (int e = 0; e < 16; ++e) b[e] = us2bf(Kp[kb_idx(e, half)]);
      acc = __builtin_amdgcn_wmma_f32_16x16x32_bf16(
          false, a, false, b, (short)0, acc, false, false);
    }
#pragma unroll
    for (int r = 0; r < 8; ++r) {
      int qrow = r + 8 * half, col = kt * 16 + l16;
      float v = acc[r] * scale;
      if (col > q0 + qrow) v = -1e9f;       // causal; exp underflows to 0
      sS[qrow * T_ + col] = v;
    }
  }
  __syncthreads();

  // ---- phase 2: row softmax in LDS, write attn to HBM (once) ----
  const int Lc = (qt + 1) * 16;
  int Lc2 = (Lc + 63) & ~63; if (Lc2 > T_) Lc2 = T_;
#pragma unroll
  for (int rr = 0; rr < 2; ++rr) {          // 8 waves x 2 rows
    int qrow = w * 2 + rr, qg = q0 + qrow;
    float* row = sS + qrow * T_;
    float m = -3.0e38f;
    for (int c = lane; c < Lc; c += 32) m = fmaxf(m, row[c]);
#pragma unroll
    for (int off = 16; off > 0; off >>= 1) m = fmaxf(m, __shfl_xor(m, off, 32));
    float s = 0.0f;
    for (int c = lane; c < Lc; c += 32) {
      float e = __expf(row[c] - m); row[c] = e; s += e;
    }
#pragma unroll
    for (int off = 16; off > 0; off >>= 1) s += __shfl_xor(s, off, 32);
    float inv = 1.0f / s;
    float* gout = attn_out + ((size_t)bh * T_ + qg) * T_;
    for (int c = lane; c < Lc; c += 32) { float p = row[c] * inv; row[c] = p; gout[c] = p; }
    for (int c = Lc + lane; c < Lc2; c += 32) row[c] = 0.0f;   // pad for WMMA
    for (int c = Lc + lane; c < T_; c += 32) gout[c] = 0.0f;   // exact zeros
  }
  __syncthreads();

  // ---- phase 3: out_head = P @ V  (split-K over 2 wave groups) ----
  // V is stored transposed [DK][T]: B-fragment elements are contiguous.
  const int nt = w & 3, ks = w >> 2;
  v8f acc;
#pragma unroll
  for (int r = 0; r < 8; ++r) acc[r] = 0.0f;
  const unsigned short* Vrow = Vt + base + (size_t)(nt * 16 + l16) * T_;
  for (int c0 = ks * 32; c0 < Lc; c0 += 64) {
    v16bf a, b;
#pragma unroll
    for (int e = 0; e < 16; ++e)
      a[e] = us2bf(f2bf(sS[l16 * T_ + c0 + ka_idx(e, half)]));
#pragma unroll
    for (int e = 0; e < 16; ++e)
      b[e] = us2bf(Vrow[c0 + kb_idx(e, half)]);
    acc = __builtin_amdgcn_wmma_f32_16x16x32_bf16(
        false, a, false, b, (short)0, acc, false, false);
  }
  if (w >= 4) {
#pragma unroll
    for (int r = 0; r < 8; ++r)
      sRed[((w - 4) * 16 + r + 8 * half) * 64 + nt * 16 + l16] = acc[r];
  }
  __syncthreads();
  if (w < 4) {
    const int bb = bh >> 4, h = bh & 15;
#pragma unroll
    for (int r = 0; r < 8; ++r) {
      float v = acc[r] + sRed[(w * 16 + r + 8 * half) * 64 + nt * 16 + l16];
      int qg = q0 + r + 8 * half, n = h * DK_ + nt * 16 + l16;
      hbuf[((size_t)bb * T_ + qg) * D_ + n] = f2bf(v);
    }
  }
}

// ---------------------------------------------------------------- launch
extern "C" void kernel_launch(void* const* d_in, const int* in_sizes, int n_in,
                              void* d_out, int out_size, void* d_ws, size_t ws_size,
                              hipStream_t stream) {
  const float* x  = (const float*)d_in[0];
  const float* Wq = (const float*)d_in[2]; const float* bq = (const float*)d_in[3];
  const float* Wk = (const float*)d_in[4]; const float* bk = (const float*)d_in[5];
  const float* Wv = (const float*)d_in[6]; const float* bv = (const float*)d_in[7];
  const float* Wo = (const float*)d_in[8]; const float* bo = (const float*)d_in[9];
  (void)in_sizes; (void)n_in; (void)out_size; (void)ws_size;

  const int NX = B_ * T_ * D_;      // 4194304
  const int NW = D_ * D_;           // 1048576
  unsigned short* ws  = (unsigned short*)d_ws;   // ~48MB used
  unsigned short* xb  = ws;
  unsigned short* Wqb = xb  + NX;
  unsigned short* Wkb = Wqb + NW;
  unsigned short* Wvb = Wkb + NW;
  unsigned short* Wob = Wvb + NW;
  unsigned short* Qh  = Wob + NW;
  unsigned short* Kh  = Qh  + NX;
  unsigned short* Vt  = Kh  + NX;
  unsigned short* hb  = Vt  + NX;

  cvt_f32_bf16<<<1024, 256, 0, stream>>>((const float4*)x,  (us4*)xb,  NX / 4);
  cvt_f32_bf16<<<512,  256, 0, stream>>>((const float4*)Wq, (us4*)Wqb, NW / 4);
  cvt_f32_bf16<<<512,  256, 0, stream>>>((const float4*)Wk, (us4*)Wkb, NW / 4);
  cvt_f32_bf16<<<512,  256, 0, stream>>>((const float4*)Wv, (us4*)Wvb, NW / 4);
  cvt_f32_bf16<<<512,  256, 0, stream>>>((const float4*)Wo, (us4*)Wob, NW / 4);

  dim3 gg(D_ / 64, (B_ * T_) / 64);               // (16, 64)
  gemm_nt<<<gg, 128, 0, stream>>>(xb, Wqb, bq, Qh, nullptr, 0, B_ * T_);
  gemm_nt<<<gg, 128, 0, stream>>>(xb, Wkb, bk, Kh, nullptr, 0, B_ * T_);
  gemm_nt<<<gg, 128, 0, stream>>>(xb, Wvb, bv, Vt, nullptr, 2, B_ * T_);

  float* out  = (float*)d_out;
  float* attn = out + (size_t)B_ * T_ * D_;
  attn_fused<<<dim3(T_ / 16, B_ * H_), 256, 0, stream>>>(Qh, Kh, Vt, hb, attn);

  gemm_nt<<<gg, 128, 0, stream>>>(hb, Wob, bo, nullptr, out, 1, B_ * T_);
}
